// GraphAttentionLayer_52432960749970
// MI455X (gfx1250) — matrix operand order repaired and checked by reading, not compiled, not run on hardware
//
#include <hip/hip_runtime.h>

// GAT layer for MI455X (gfx1250, wave32, WMMA, TDM).
// N=6144, INPUT=1024, HIDDEN=256.
#define NN   6144
#define KIN  1024
#define HID  256
#define NEGINF (-9.0e15f)
#define NCH  (NN / 128)   // 48 adjacency chunks per row
#define NBW  (NN / 32)    // 192 bitmask words per row (4 per chunk)

typedef __attribute__((ext_vector_type(2)))  float    v2f;
typedef __attribute__((ext_vector_type(4)))  float    v4f;
typedef __attribute__((ext_vector_type(8)))  float    v8f;
typedef __attribute__((ext_vector_type(4)))  int      v4i;
typedef __attribute__((ext_vector_type(4)))  unsigned v4u;
typedef __attribute__((ext_vector_type(8)))  __bf16   v8bf;
typedef __attribute__((ext_vector_type(16))) __bf16   v16bf;
typedef __attribute__((ext_vector_type(4)))  unsigned u32x4;
typedef __attribute__((ext_vector_type(4)))  int      i32x4;
typedef __attribute__((ext_vector_type(8)))  int      i32x8;

#if __has_builtin(__builtin_amdgcn_tensor_load_to_lds) && \
    __has_builtin(__builtin_amdgcn_s_wait_tensorcnt)
#define USE_TDM 1
#endif

// ---------------------------------------------------------------------------
// Bitmask layout (producer k_stats / consumer k_attn agree):
//   column c -> word (c>>7)*4 + (c&3), bit (c>>2)&31
// (comes from b128 adjacency loads: lane L owns columns 4L..4L+3 of a
//  128-column chunk; ballot t packs bit t of every lane's quad)
// ---------------------------------------------------------------------------

// ---------------------------------------------------------------------------
// K1: h = X @ W (fp32 WMMA 16x16x4). One 16x16 tile per wave.
// Emits h fp32 (for s-vectors) and h in *pre-swizzled bf16 B-fragment order*:
//   hbp[((row>>5)*HID + col)*32 + (row&31)]
// so each 32-row K-tile of B is a contiguous 16 KB block (TDM/vector friendly).
// ---------------------------------------------------------------------------
__global__ __launch_bounds__(256) void k_gemm1(const float* __restrict__ X,
                                               const float* __restrict__ W,
                                               float* __restrict__ h,
                                               __bf16* __restrict__ hbp) {
    const int lane = threadIdx.x & 31;
    const int wave = threadIdx.x >> 5;
    const int tile = blockIdx.x * 8 + wave;   // 0..6143
    const int tm   = tile >> 4;               // M tile 0..383
    const int tn   = tile & 15;               // N tile 0..15
    const int ml   = lane & 15;
    const int hi   = lane >> 4;

    v8f acc = {};
#if __has_builtin(__builtin_amdgcn_wmma_f32_16x16x4_f32)
    // f32 A 16x4: lanes 0-15 K=0,1 ; lanes 16-31 K=2,3
    const float* arow = X + (size_t)(tm * 16 + ml) * KIN + hi * 2;
    const float* bcol = W + tn * 16 + ml + (size_t)(hi * 2) * HID;
    for (int k = 0; k < KIN; k += 4) {
        v2f a = *(const v2f*)(arow + k);
        v2f b;
        b.x = bcol[(size_t)k * HID];
        b.y = bcol[(size_t)k * HID + HID];
        acc = __builtin_amdgcn_wmma_f32_16x16x4_f32(false, a, false, b,
                                                    (short)0, acc, false, false);
    }
#else
    const int m_g = tm * 16 + ml;
    const int n_g = tn * 16 + ml;
    for (int k = 0; k < KIN; k += 32) {
        v16bf a, b;
#pragma unroll
        for (int i = 0; i < 8; ++i) {
            a[i]     = (__bf16)X[(size_t)m_g * KIN + k + hi * 8 + i];
            a[8 + i] = (__bf16)X[(size_t)m_g * KIN + k + 16 + hi * 8 + i];
        }
#pragma unroll
        for (int i = 0; i < 16; ++i)
            b[i] = (__bf16)W[(size_t)(k + hi * 16 + i) * HID + n_g];
        acc = __builtin_amdgcn_wmma_f32_16x16x32_bf16(false, a, false, b,
                                                      (short)0, acc, false, false);
    }
#endif
#pragma unroll
    for (int r = 0; r < 8; ++r) {
        const int row = tm * 16 + r + 8 * hi;
        const int col = tn * 16 + ml;
        const float v = acc[r];
        h[(size_t)row * HID + col] = v;
        hbp[((size_t)(row >> 5) * HID + col) * 32 + (row & 31)] = (__bf16)v;
    }
}

// ---------------------------------------------------------------------------
// K2: s1 = h @ a1, s2 = h @ a2 (wave-per-row, wave32 shuffle reduction)
// ---------------------------------------------------------------------------
__global__ __launch_bounds__(256) void k_svec(const float* __restrict__ h,
                                              const float* __restrict__ a_edge,
                                              float* __restrict__ s1,
                                              float* __restrict__ s2) {
    const int lane = threadIdx.x & 31;
    const int row  = blockIdx.x * 8 + (threadIdx.x >> 5);
    float p1 = 0.f, p2 = 0.f;
    for (int c = lane; c < HID; c += 32) {
        const float hv = h[(size_t)row * HID + c];
        p1 += hv * a_edge[c];
        p2 += hv * a_edge[HID + c];
    }
#pragma unroll
    for (int off = 16; off > 0; off >>= 1) {
        p1 += __shfl_xor(p1, off, 32);
        p2 += __shfl_xor(p2, off, 32);
    }
    if (lane == 0) { s1[row] = p1; s2[row] = p2; }
}

// ---------------------------------------------------------------------------
// K3: single HBM pass over adj (151 MB = the roofline floor), b128 loads:
//   - pack adjacency to 4.7 MB bitmask (ballot_w32 x4 per 128-col chunk)
//   - online softmax stats (row max, row sum-of-exp) with exact NEGINF path
// ---------------------------------------------------------------------------
__global__ __launch_bounds__(256) void k_stats(const int* __restrict__ adj,
                                               const float* __restrict__ s1,
                                               const float* __restrict__ s2,
                                               unsigned* __restrict__ bits,
                                               float* __restrict__ rowm,
                                               float* __restrict__ rowl) {
    const int lane = threadIdx.x & 31;
    const int row  = blockIdx.x * 8 + (threadIdx.x >> 5);
    const int* arow = adj + (size_t)row * NN;
    const float s1v = s1[row];
    float m = -INFINITY, l = 0.f;
    for (int ch = 0; ch < NCH; ++ch) {
        const int c0 = ch * 128 + lane * 4;
        if (ch + 16 < NCH)
            __builtin_prefetch(arow + c0 + 16 * 128, 0, 0);  // global_prefetch
        const v4i av = *(const v4i*)(arow + c0);
        const v4f sv = *(const v4f*)(s2 + c0);
        float w[4];
        unsigned msk[4];
#pragma unroll
        for (int t = 0; t < 4; ++t) {
            const float sc = s1v + sv[t];
            w[t] = (av[t] != 0) ? (sc > 0.f ? sc : 0.2f * sc) : NEGINF;
#if __has_builtin(__builtin_amdgcn_ballot_w32)
            msk[t] = __builtin_amdgcn_ballot_w32(av[t] != 0);
#else
            msk[t] = (unsigned)__ballot(av[t] != 0);
#endif
        }
        if (lane == 0) {
            v4u wv = {msk[0], msk[1], msk[2], msk[3]};
            *(v4u*)(bits + (size_t)row * NBW + ch * 4) = wv;
        }
        const float m4 = fmaxf(fmaxf(w[0], w[1]), fmaxf(w[2], w[3]));
        const float l4 = __expf(w[0] - m4) + __expf(w[1] - m4) +
                         __expf(w[2] - m4) + __expf(w[3] - m4);
        const float nm = fmaxf(m, m4);
        l = l * __expf(m - nm) + l4 * __expf(m4 - nm);
        m = nm;
    }
#pragma unroll
    for (int off = 16; off > 0; off >>= 1) {
        const float mo = __shfl_xor(m, off, 32);
        const float lo = __shfl_xor(l, off, 32);
        const float nm = fmaxf(m, mo);
        l = l * __expf(m - nm) + lo * __expf(mo - nm);
        m = nm;
    }
    if (lane == 0) { rowm[row] = m; rowl[row] = l; }
}

// ---------------------------------------------------------------------------
// TDM helper: 1-D tile load of one contiguous 16 KB B-chunk into LDS.
// D# group0: count=1, lds_addr, 57-bit global_addr, type=2.
// D# group1: data_size=8B, tensor_dim0=2^30, tensor_dim1=1, tile_dim0=2048.
// ---------------------------------------------------------------------------
#ifdef USE_TDM
__device__ __forceinline__ void tdm_load_16k(const __bf16* src, unsigned ldsOff) {
    const unsigned long long ga = (unsigned long long)(uintptr_t)src;
    u32x4 g0 = { 1u, ldsOff, (unsigned)ga,
                 (unsigned)((ga >> 32) & 0x01FFFFFFu) | (2u << 30) };
    i32x8 g1 = { (int)(3u << 16),      // data_size=3 (8B)
                 0,                    // atomic barrier addr = 0
                 (int)0x00014000,      // tensor_dim0 hi16=0x4000, tensor_dim1 lo16=1
                 (int)0x08000000,      // tile_dim0 = 2048 (<<16)
                 0,                    // tile_dim1/2 = 0 (1-D tile)
                 (int)0x40000000,      // tensor_dim0_stride lo32
                 0, 0 };
    i32x4 z4 = { 0, 0, 0, 0 };
#if defined(__clang_major__) && (__clang_major__ >= 23)
    i32x8 z8 = { 0, 0, 0, 0, 0, 0, 0, 0 };
    __builtin_amdgcn_tensor_load_to_lds(g0, g1, z4, z4, z8, 0);
#else
    __builtin_amdgcn_tensor_load_to_lds(g0, g1, z4, z4, 0);
#endif
}
#endif

// ---------------------------------------------------------------------------
// K4: out = softmax(weight) @ h, bf16 WMMA with fp32 accumulate.
// One block per 16-row slab. Per K-step of 32:
//   - block materializes P[16][32] in LDS from bitmask + final row stats
//   - B tile (contiguous 16 KB of pre-swizzled h) double-buffered into LDS
//     by wave 0 via TENSOR_LOAD_TO_LDS (+ s_wait_tensorcnt), fallback: direct
//     32-byte vector loads from L2-resident hbp
//   - each wave: A from LDS (ds_load_b128 pairs), 2x v_wmma_f32_16x16x32_bf16
// Stats are final -> accumulators never rescale.
// ---------------------------------------------------------------------------
__global__ __launch_bounds__(256) void k_attn(const unsigned* __restrict__ bits,
                                              const float* __restrict__ s1,
                                              const float* __restrict__ s2,
                                              const float* __restrict__ rowm,
                                              const float* __restrict__ rowl,
                                              const __bf16* __restrict__ hbp,
                                              float* __restrict__ out) {
    __shared__ __align__(16) __bf16 pT[16 * 32];
#ifdef USE_TDM
    __shared__ __align__(64) __bf16 bT[2][8192];   // 2 x 16 KB B staging
#endif
    const int lane = threadIdx.x & 31;
    const int wave = threadIdx.x >> 5;
    const int i0   = blockIdx.x * 16;

    // producer mapping: thread -> (row pm, even k-pair pk)
    const int pm = threadIdx.x & 15;
    const int pk = (threadIdx.x >> 4) * 2;
    const float s1v = s1[i0 + pm];
    const float mv  = rowm[i0 + pm];
    const float il  = 1.0f / rowl[i0 + pm];
    const unsigned* brow = bits + (size_t)(i0 + pm) * NBW;

    // consumer mapping (WMMA fragments)
    const int ml    = lane & 15;
    const int hi    = lane >> 4;
    const int n0    = wave * 32 + ml;   // this wave's column strip
    const int koffA = hi * 16;
    v8f acc0 = {}, acc1 = {};

#ifdef USE_TDM
    const unsigned ldsOff[2] = { (unsigned)(uintptr_t)&bT[0][0],
                                 (unsigned)(uintptr_t)&bT[1][0] };
    if (wave == 0) tdm_load_16k(hbp, ldsOff[0]);   // prologue: K-tile 0
#endif

    for (int jb = 0; jb < NBW; ++jb) {
        // --- produce P[16][32] in LDS ---
#pragma unroll
        for (int t = 0; t < 2; ++t) {
            const int c = jb * 32 + pk + t;                      // global column
            const unsigned word = brow[((c >> 7) << 2) + (c & 3)];
            const unsigned on   = (word >> ((c >> 2) & 31)) & 1u;
            const float sc = s1v + s2[c];
            const float wv = on ? (sc > 0.f ? sc : 0.2f * sc) : NEGINF;
            pT[pm * 32 + pk + t] = (__bf16)(__expf(wv - mv) * il);
        }
#ifdef USE_TDM
        if (wave == 0) __builtin_amdgcn_s_wait_tensorcnt(0);     // buf jb&1 ready
#endif
        __syncthreads();
#ifdef USE_TDM
        if (wave == 0 && jb + 1 < NBW)
            tdm_load_16k(hbp + (size_t)(jb + 1) * 8192, ldsOff[(jb + 1) & 1]);
#endif
        // --- A fragment (16x32 bf16) from LDS ---
        const __bf16* pr = &pT[ml * 32];
        const v8bf alo = *(const v8bf*)(pr + koffA);
        const v8bf ahi = *(const v8bf*)(pr + 8 + koffA);
        const v16bf a = __builtin_shufflevector(alo, ahi,
                            0, 1, 2, 3, 4, 5, 6, 7, 8, 9, 10, 11, 12, 13, 14, 15);
        // --- B fragments (32-byte contiguous per lane, pre-swizzled) ---
#ifdef USE_TDM
        const __bf16* bbase = &bT[jb & 1][0];
        const v16bf b0 = *(const v16bf*)(bbase + (unsigned)(n0)      * 32 + hi * 16);
        const v16bf b1 = *(const v16bf*)(bbase + (unsigned)(n0 + 16) * 32 + hi * 16);
#else
        const __bf16* bbase = hbp + (size_t)jb * 8192;
        const v16bf b0 = *(const v16bf*)(bbase + (size_t)(n0)      * 32 + hi * 16);
        const v16bf b1 = *(const v16bf*)(bbase + (size_t)(n0 + 16) * 32 + hi * 16);
#endif
        acc0 = __builtin_amdgcn_wmma_f32_16x16x32_bf16(false, a, false, b0,
                                                       (short)0, acc0, false, false);
        acc1 = __builtin_amdgcn_wmma_f32_16x16x32_bf16(false, a, false, b1,
                                                       (short)0, acc1, false, false);
        __syncthreads();
    }
#pragma unroll
    for (int r = 0; r < 8; ++r) {
        const int row = i0 + r + 8 * hi;
        out[(size_t)row * HID + n0]      = acc0[r];
        out[(size_t)row * HID + n0 + 16] = acc1[r];
    }
}

// ---------------------------------------------------------------------------
extern "C" void kernel_launch(void* const* d_in, const int* in_sizes, int n_in,
                              void* d_out, int out_size, void* d_ws, size_t ws_size,
                              hipStream_t stream) {
    (void)in_sizes; (void)n_in; (void)out_size; (void)ws_size;
    const float* X      = (const float*)d_in[0];
    const float* W      = (const float*)d_in[1];
    const float* a_edge = (const float*)d_in[2];
    const int*   adj    = (const int*)d_in[3];
    float* out = (float*)d_out;

    char* ws = (char*)d_ws;
    auto carve = [&](size_t bytes) -> void* {
        void* p = ws;
        ws += (bytes + 255) & ~(size_t)255;
        return p;
    };
    float*    h    = (float*)carve((size_t)NN * HID * 4);
    __bf16*   hbp  = (__bf16*)carve((size_t)NN * HID * 2);  // pre-swizzled bf16
    float*    s1   = (float*)carve((size_t)NN * 4);
    float*    s2   = (float*)carve((size_t)NN * 4);
    float*    rowm = (float*)carve((size_t)NN * 4);
    float*    rowl = (float*)carve((size_t)NN * 4);
    unsigned* bitm = (unsigned*)carve((size_t)NN * NBW * 4);

    k_gemm1<<<(NN / 16) * (HID / 16) / 8, 256, 0, stream>>>(X, W, h, hbp);    // 768
    k_svec <<<NN / 8,  256, 0, stream>>>(h, a_edge, s1, s2);                  // 768
    k_stats<<<NN / 8,  256, 0, stream>>>(adj, s1, s2, bitm, rowm, rowl);      // 768
    k_attn <<<NN / 16, 256, 0, stream>>>(bitm, s1, s2, rowm, rowl, hbp, out); // 384
}
